// EST_40072044872217
// MI455X (gfx1250) — compile-verified, matrix operand order checked by reading, not run on hardware
//
#include <hip/hip_runtime.h>
#include <hip/hip_bf16.h>
#include <math.h>

// Problem constants (from reference)
#define BB 64      // batch
#define UU 8       // units
#define NN 2048    // reservoir size
#define DD 512     // input dim
#define OO 512     // output dim

typedef __attribute__((ext_vector_type(16))) __bf16 v16bf;
typedef __attribute__((ext_vector_type(8)))  float  v8f;

// ---------------------------------------------------------------------------
// gfx1250 async global->LDS copy (ASYNCcnt tracked). VDST = LDS byte address,
// VADDR = 64-bit global address. Inline asm per CDNA5 ISA 15.18.3 (opcode 98).
// ---------------------------------------------------------------------------
__device__ __forceinline__ void async_b128(const float* g, float* lds) {
    unsigned loff = (unsigned)(unsigned long long)lds;   // low 32b = LDS offset
    asm volatile("global_load_async_to_lds_b128 %0, %1, off"
                 :: "v"(loff), "v"(g) : "memory");
}
__device__ __forceinline__ void wait_async0() {
    asm volatile("s_wait_asynccnt 0" ::: "memory");
}

// branchless tanh: one v_exp_f32, no exec-mask divergence
__device__ __forceinline__ float fast_tanh(float x) {
    const float t = __expf(-2.f * fabsf(x));
    return copysignf((1.f - t) / (1.f + t), x);
}

// ---------------------------------------------------------------------------
// Kernel 1: logits -> softmax over U  ->  lr_ws[b*U + u]
// ---------------------------------------------------------------------------
__global__ __launch_bounds__(256) void lr_kernel(const float* __restrict__ X,
                                                 const float* __restrict__ alr,
                                                 const float* __restrict__ temp,
                                                 float* __restrict__ lr_ws) {
    const int b    = blockIdx.x;
    const int w    = threadIdx.x >> 5;   // unit
    const int lane = threadIdx.x & 31;
    __shared__ float logits[UU];

    const float* x = X + (b * UU + w) * DD;
    const float* a = alr + w * DD;
    float s = 0.f;
    #pragma unroll 4
    for (int d = lane; d < DD; d += 32) s += x[d] * a[d];
    #pragma unroll
    for (int off = 16; off > 0; off >>= 1) s += __shfl_xor(s, off, 32);
    if (lane == 0) logits[w] = s / temp[0];
    __syncthreads();

    if (threadIdx.x < UU) {
        float m = logits[0];
        #pragma unroll
        for (int i = 1; i < UU; ++i) m = fmaxf(m, logits[i]);
        float sum = 0.f;
        #pragma unroll
        for (int i = 0; i < UU; ++i) sum += __expf(logits[i] - m);
        lr_ws[b * UU + threadIdx.x] = __expf(logits[threadIdx.x] - m) / sum;
    }
}

// ---------------------------------------------------------------------------
// WMMA fragment layouts (wave32, CDNA5 ISA 7.12.2)
//   A (16-bit 16x32): lane -> m = L&15, half = L>>4;
//     e<8: K = 8*half + e ; e>=8: K = 16 + 8*half + (e-8)
//   B (32x16): lane -> n = L&15, same K walk
//   C/D f32:   VGPR r -> M = r + 8*half, N = L&15
// A tile stored FRAGMENT-MAJOR in LDS: Af[mtile][lane][e] (32B/lane,
// contiguous) -> fragment load = 2x ds_load_b128, no address math.
// ---------------------------------------------------------------------------
union BFrag  { v16bf v; __bf16 e[16]; };
union Pack8  { __bf16 e[8]; uint4 q; };

// ---------------------------------------------------------------------------
// Kernel 2: new_state = (1-lr)*state + lr*tanh(X@Win + state@(W*sr) + bias)
// grid (NN/128, UU), 256 threads (8 waves); M = 64 (full batch), 128 cols/blk.
// Fused K = 512 (Win, A=X) + 2048 (W, A=state), split into two phase loops so
// the hot W loop has no per-element selects. B tile (32x128 f32, 16KB)
// streamed via async global->LDS, double buffered across the phase boundary.
// ---------------------------------------------------------------------------
__global__ __launch_bounds__(256) void state_kernel(const float* __restrict__ X,
                                                    const float* __restrict__ state,
                                                    const float* __restrict__ W,
                                                    const float* __restrict__ Win,
                                                    const float* __restrict__ bias,
                                                    const float* __restrict__ sr,
                                                    const float* __restrict__ lr_ws,
                                                    float* __restrict__ new_state) {
    const int u    = blockIdx.y;
    const int nb0  = blockIdx.x * 128;
    const int wav  = threadIdx.x >> 5;
    const int lane = threadIdx.x & 31;
    const int half = lane >> 4;
    const int n    = nb0 + wav * 16 + (lane & 15);
    const int nloc = wav * 16 + (lane & 15);

    __shared__ __align__(16) float  Bls[2][32][128];    // 2 x 16 KB, async-filled
    __shared__ __align__(32) __bf16 Af[2][4][32][16];   // 2 x 4 KB, fragment-major

    v8f acc[4] = {v8f{}, v8f{}, v8f{}, v8f{}};
    const float srv = sr[u];

    // A staging map: 256 threads = 64 rows x 4, 8 floats each.
    const int ar = threadIdx.x >> 2;
    const int ak = (threadIdx.x & 3) * 8;
    const int amt   = ar >> 4;
    const int alane = (ar & 15) + 16 * ((ak >> 3) & 1);
    const int ae0   = (ak & 16) >> 1;
    const float* xrow = X     + (size_t)(ar * UU + u) * DD;
    const float* srow = state + (size_t)(ar * UU + u) * NN;
    const float* Wu   = W     + (size_t)u * NN * NN;
    const float* Winu = Win   + (size_t)u * DD * NN;

    auto issueB = [&](const float* Bsrc, int k0, int buf) {  // 1024 x 16B chunks
        #pragma unroll
        for (int i = 0; i < 4; ++i) {
            const int c   = threadIdx.x + 256 * i;
            const int row = c >> 5;        // 32 chunks per 128-float row
            const int cc  = c & 31;
            async_b128(Bsrc + (size_t)(k0 + row) * NN + nb0 + cc * 4,
                       &Bls[buf][row][cc * 4]);
        }
    };
    auto stageA = [&](const float* rowp, int k0, int buf) {  // f32->bf16, 1 b128
        const float* src = rowp + k0 + ak;
        float4 f0 = *(const float4*)(src);
        float4 f1 = *(const float4*)(src + 4);
        Pack8 p;
        p.e[0] = (__bf16)f0.x; p.e[1] = (__bf16)f0.y;
        p.e[2] = (__bf16)f0.z; p.e[3] = (__bf16)f0.w;
        p.e[4] = (__bf16)f1.x; p.e[5] = (__bf16)f1.y;
        p.e[6] = (__bf16)f1.z; p.e[7] = (__bf16)f1.w;
        *(uint4*)&Af[buf][amt][alane][ae0] = p.q;
    };
    auto compute = [&](int cur, bool scale) {   // scale folds at call site
        BFrag bfr;
        #pragma unroll
        for (int e = 0; e < 16; ++e) {
            const int k = ((e < 8) ? e : e + 8) + 8 * half;
            float f = Bls[cur][k][nloc];
            if (scale) f *= srv;
            bfr.e[e] = (__bf16)f;
        }
        v16bf af[4];
        #pragma unroll
        for (int mt = 0; mt < 4; ++mt)
            af[mt] = *(const v16bf*)&Af[cur][mt][lane][0];
        #pragma unroll
        for (int mt = 0; mt < 4; ++mt)
            acc[mt] = __builtin_amdgcn_wmma_f32_16x16x32_bf16(
                false, af[mt], false, bfr.v, (short)0, acc[mt], false, false);
    };

    // prologue
    issueB(Winu, 0, 0);
    stageA(xrow, 0, 0);
    wait_async0();
    __syncthreads();

    // phase 0: B = Win, A = X, 16 steps of K=32, no scaling
    for (int kt = 0; kt < 16; ++kt) {
        const int cur = kt & 1, nxt = cur ^ 1;
        if (kt < 15) { issueB(Winu, (kt + 1) * 32, nxt); stageA(xrow, (kt + 1) * 32, nxt); }
        else         { issueB(Wu, 0, nxt);               stageA(srow, 0, nxt); }
        compute(cur, false);
        wait_async0();
        __syncthreads();
    }
    // phase 1: B = W (x sr), A = state, 64 steps of K=32
    for (int kt = 0; kt < 64; ++kt) {
        const int cur = kt & 1, nxt = cur ^ 1;   // (16+kt)&1 == kt&1
        if (kt + 1 < 64) { issueB(Wu, (kt + 1) * 32, nxt); stageA(srow, (kt + 1) * 32, nxt); }
        compute(cur, true);
        wait_async0();
        __syncthreads();
    }

    // epilogue: bias + tanh + lr blend, write new_state into d_out
    const float bs = bias[u * NN + n];
    #pragma unroll
    for (int mt = 0; mt < 4; ++mt) {
        #pragma unroll
        for (int r = 0; r < 8; ++r) {
            const int m = mt * 16 + r + 8 * half;     // batch index
            const float lr = lr_ws[m * UU + u];
            const float h  = fast_tanh(acc[mt][r] + bs);
            const float st = state[(size_t)(m * UU + u) * NN + n];
            new_state[(size_t)(m * UU + u) * NN + n] = (1.f - lr) * st + lr * h;
        }
    }
}

// ---------------------------------------------------------------------------
// Kernel 3: output = new_state @ Wout   (per u: 64x2048 @ 2048x512)
// grid (OO/128, UU); K = 2048 -> 64 steps of 32. Same async-pipelined scheme.
// ---------------------------------------------------------------------------
__global__ __launch_bounds__(256) void out_kernel(const float* __restrict__ new_state,
                                                  const float* __restrict__ Wout,
                                                  float* __restrict__ out) {
    const int u    = blockIdx.y;
    const int nb0  = blockIdx.x * 128;
    const int wav  = threadIdx.x >> 5;
    const int lane = threadIdx.x & 31;
    const int half = lane >> 4;
    const int n    = nb0 + wav * 16 + (lane & 15);
    const int nloc = wav * 16 + (lane & 15);

    __shared__ __align__(16) float  Bls[2][32][128];
    __shared__ __align__(32) __bf16 Af[2][4][32][16];

    v8f acc[4] = {v8f{}, v8f{}, v8f{}, v8f{}};

    const int ar = threadIdx.x >> 2;
    const int ak = (threadIdx.x & 3) * 8;
    const int amt   = ar >> 4;
    const int alane = (ar & 15) + 16 * ((ak >> 3) & 1);
    const int ae0   = (ak & 16) >> 1;
    const float* srow  = new_state + (size_t)(ar * UU + u) * NN;
    const float* Woutu = Wout + (size_t)u * NN * OO;

    auto issueB = [&](int k0, int buf) {
        #pragma unroll
        for (int i = 0; i < 4; ++i) {
            const int c   = threadIdx.x + 256 * i;
            const int row = c >> 5;
            const int cc  = c & 31;
            async_b128(Woutu + (size_t)(k0 + row) * OO + nb0 + cc * 4,
                       &Bls[buf][row][cc * 4]);
        }
    };
    auto stageA = [&](int k0, int buf) {
        const float* src = srow + k0 + ak;
        float4 f0 = *(const float4*)(src);
        float4 f1 = *(const float4*)(src + 4);
        Pack8 p;
        p.e[0] = (__bf16)f0.x; p.e[1] = (__bf16)f0.y;
        p.e[2] = (__bf16)f0.z; p.e[3] = (__bf16)f0.w;
        p.e[4] = (__bf16)f1.x; p.e[5] = (__bf16)f1.y;
        p.e[6] = (__bf16)f1.z; p.e[7] = (__bf16)f1.w;
        *(uint4*)&Af[buf][amt][alane][ae0] = p.q;
    };

    issueB(0, 0);
    stageA(0, 0);
    wait_async0();
    __syncthreads();

    for (int kt = 0; kt < 64; ++kt) {
        const int cur = kt & 1, nxt = cur ^ 1;
        if (kt + 1 < 64) {
            issueB((kt + 1) * 32, nxt);
            stageA((kt + 1) * 32, nxt);
        }

        BFrag bfr;
        #pragma unroll
        for (int e = 0; e < 16; ++e) {
            const int k = ((e < 8) ? e : e + 8) + 8 * half;
            bfr.e[e] = (__bf16)Bls[cur][k][nloc];
        }
        v16bf af[4];
        #pragma unroll
        for (int mt = 0; mt < 4; ++mt)
            af[mt] = *(const v16bf*)&Af[cur][mt][lane][0];
        #pragma unroll
        for (int mt = 0; mt < 4; ++mt)
            acc[mt] = __builtin_amdgcn_wmma_f32_16x16x32_bf16(
                false, af[mt], false, bfr.v, (short)0, acc[mt], false, false);

        wait_async0();
        __syncthreads();
    }

    #pragma unroll
    for (int mt = 0; mt < 4; ++mt) {
        #pragma unroll
        for (int r = 0; r < 8; ++r) {
            const int m = mt * 16 + r + 8 * half;
            out[(size_t)(m * UU + u) * OO + n] = acc[mt][r];
        }
    }
}

// ---------------------------------------------------------------------------
extern "C" void kernel_launch(void* const* d_in, const int* in_sizes, int n_in,
                              void* d_out, int out_size, void* d_ws, size_t ws_size,
                              hipStream_t stream) {
    const float* X     = (const float*)d_in[0];
    const float* state = (const float*)d_in[1];
    const float* W     = (const float*)d_in[2];
    const float* Win   = (const float*)d_in[3];
    const float* bias  = (const float*)d_in[4];
    const float* Wout  = (const float*)d_in[5];
    const float* sr    = (const float*)d_in[6];
    const float* alr   = (const float*)d_in[7];
    const float* temp  = (const float*)d_in[8];

    float* out   = (float*)d_out;                 // new_state first
    float* out2  = out + (size_t)BB * UU * NN;    // then output
    float* lr_ws = (float*)d_ws;                  // BB*UU floats

    lr_kernel<<<BB, 256, 0, stream>>>(X, alr, temp, lr_ws);
    state_kernel<<<dim3(NN / 128, UU), 256, 0, stream>>>(X, state, W, Win, bias,
                                                         sr, lr_ws, out);
    out_kernel<<<dim3(OO / 128, UU), 256, 0, stream>>>(out, Wout, out2);
}
